// PatientGCN_45861660786779
// MI455X (gfx1250) — matrix operand order repaired and checked by reading, not compiled, not run on hardware
//
#include <hip/hip_runtime.h>
#include <hip/hip_bf16.h>

typedef __attribute__((ext_vector_type(16))) __bf16 v16bf;
typedef __attribute__((ext_vector_type(8)))  __bf16 v8bf;
typedef __attribute__((ext_vector_type(8)))  float  v8f;

#define FDIM 128
#define WPITCH 136   // LDS row pitch (bf16 elems) for transposed W: 272B rows -> conflict-free

// ---------------- degree / normalization ----------------
__global__ void deg_init_kernel(float* __restrict__ deg, int n) {
    int i = blockIdx.x * blockDim.x + threadIdx.x;
    if (i < n) deg[i] = 1.0f;   // self loop
}

__global__ void deg_count_kernel(const int* __restrict__ dst, float* __restrict__ deg, int e) {
    int i = blockIdx.x * blockDim.x + threadIdx.x;
    if (i < e) atomicAdd(&deg[dst[i]], 1.0f);
}

__global__ void deg_rsqrt_kernel(float* __restrict__ dis, int n) {
    int i = blockIdx.x * blockDim.x + threadIdx.x;
    if (i < n) dis[i] = __frsqrt_rn(dis[i]);
}

// ---------------- GEMM: T = Hin @ W  (bf16 WMMA, f32 accum) ----------------
// One wave computes a 16x128 output strip: K=128 in 4 steps of 32, N=128 as 8 tiles.
__global__ __launch_bounds__(256) void gemm_bf16_kernel(const float* __restrict__ Hin,
                                                        const float* __restrict__ W,
                                                        float* __restrict__ Tout,
                                                        int nrows) {
    __shared__ __bf16 Wt[FDIM * WPITCH];   // transposed: Wt[n*WPITCH + k], 34.8 KB
    int tid = threadIdx.x;
    for (int i = tid; i < FDIM * FDIM; i += 256) {
        int k = i >> 7, n = i & 127;
        Wt[n * WPITCH + k] = (__bf16)W[i];   // W is [K=128][N=128] row-major
    }
    __syncthreads();

    int wave = tid >> 5;
    int lane = tid & 31;
    int row_tile = (blockIdx.x * 8 + wave) * 16;
    if (row_tile >= nrows) return;          // wave-uniform; EXEC stays all-1s for WMMA

    // A 16x32 bf16, this code's contiguous k order: lanes 0-15 -> M=lane, K 0..15; lanes 16-31 -> K 16..31
    int mrow  = row_tile + (lane & 15);
    int koff8 = (lane >> 4) * 16;
    // B 32x16 bf16, same contiguous k order as A: lanes 0-15 -> N=lane, K 0..15; lanes 16-31 -> K 16..31
    int bn    = lane & 15;
    int bkoff = (lane >> 4) * 16;

    v8f acc[8] = {};
    const float* hrow = Hin + (size_t)mrow * FDIM;

    #pragma unroll
    for (int ks = 0; ks < 4; ++ks) {
        int kbase = ks * 32;
        const float4* ap0 = (const float4*)(hrow + kbase + koff8);       // K+0..7  (32B aligned)
        const float4* ap1 = (const float4*)(hrow + kbase + koff8 + 8);   // K+8..15
        float4 a0 = ap0[0], a1 = ap0[1], a2 = ap1[0], a3 = ap1[1];
        v16bf a;
        a[0]=(__bf16)a0.x;  a[1]=(__bf16)a0.y;  a[2]=(__bf16)a0.z;  a[3]=(__bf16)a0.w;
        a[4]=(__bf16)a1.x;  a[5]=(__bf16)a1.y;  a[6]=(__bf16)a1.z;  a[7]=(__bf16)a1.w;
        a[8]=(__bf16)a2.x;  a[9]=(__bf16)a2.y;  a[10]=(__bf16)a2.z; a[11]=(__bf16)a2.w;
        a[12]=(__bf16)a3.x; a[13]=(__bf16)a3.y; a[14]=(__bf16)a3.z; a[15]=(__bf16)a3.w;

        #pragma unroll
        for (int nt = 0; nt < 8; ++nt) {
            const v8bf* bp = (const v8bf*)&Wt[(size_t)(nt * 16 + bn) * WPITCH + kbase + bkoff];
            v8bf blo = bp[0], bhi = bp[1];   // 16 consecutive bf16, 16B-aligned
            v16bf b = __builtin_shufflevector(blo, bhi,
                                              0,1,2,3,4,5,6,7,8,9,10,11,12,13,14,15);
            acc[nt] = __builtin_amdgcn_wmma_f32_16x16x32_bf16(
                          false, a, false, b, (short)0, acc[nt], false, false);
        }
    }

    // C/D layout: VGPR r: lanes 0-15 -> M=r, N=lane; lanes 16-31 -> M=r+8, N=lane-16
    int srow  = row_tile + (lane >> 4) * 8;
    int scol0 = lane & 15;
    #pragma unroll
    for (int nt = 0; nt < 8; ++nt) {
        #pragma unroll
        for (int r = 0; r < 8; ++r) {
            Tout[(size_t)(srow + r) * FDIM + nt * 16 + scol0] = acc[nt][r];
        }
    }
}

// ---------------- agg = t[i] * dis[i]^2   (self-loop message) ----------------
__global__ void self_init_kernel(const float* __restrict__ T, const float* __restrict__ dis,
                                 float* __restrict__ agg, int n) {
    int tid = blockIdx.x * blockDim.x + threadIdx.x;
    int node = tid >> 5;
    if (node >= n) return;
    int l = tid & 31;
    float di = dis[node];
    float w = di * di;
    const float4* tp = (const float4*)(T + (size_t)node * FDIM + l * 4);
    float4 v = *tp;
    v.x *= w; v.y *= w; v.z *= w; v.w *= w;
    *(float4*)(agg + (size_t)node * FDIM + l * 4) = v;
}

// ---------------- edge scatter: agg[dst] += t[src] * dis[src]*dis[dst] ----------------
__global__ void edge_scatter_kernel(const int* __restrict__ src, const int* __restrict__ dst,
                                    const float* __restrict__ dis, const float* __restrict__ T,
                                    float* __restrict__ agg, int e) {
    int tid = blockIdx.x * blockDim.x + threadIdx.x;
    int ed = tid >> 5;
    if (ed >= e) return;
    int l = tid & 31;
    int s = src[ed], d = dst[ed];
    float w = dis[s] * dis[d];
    float4 v = *(const float4*)(T + (size_t)s * FDIM + l * 4);
    float* ap = agg + (size_t)d * FDIM + l * 4;
    atomicAdd(ap + 0, v.x * w);
    atomicAdd(ap + 1, v.y * w);
    atomicAdd(ap + 2, v.z * w);
    atomicAdd(ap + 3, v.w * w);
}

// ---------------- h = relu(agg + b) ----------------
__global__ void bias_relu_kernel(const float* __restrict__ agg, const float* __restrict__ bias,
                                 float* __restrict__ hout, int total) {
    int tid = blockIdx.x * blockDim.x + threadIdx.x;
    if (tid >= total) return;
    int c = tid & 127;
    hout[tid] = fmaxf(agg[tid] + bias[c], 0.0f);
}

// ---------------- graph max pool ----------------
__global__ void pool_init_kernel(unsigned* __restrict__ gmax) {
    gmax[threadIdx.x] = 0u;   // +0.0f bits; valid identity since h >= 0 after relu
}

__global__ void pool_max_kernel(const float* __restrict__ h, unsigned* __restrict__ gmax, int n) {
    int c = threadIdx.x;                 // 128 threads
    int r0 = blockIdx.x * 128;
    int r1 = r0 + 128; if (r1 > n) r1 = n;
    float m = 0.0f;
    for (int r = r0; r < r1; ++r) m = fmaxf(m, h[(size_t)r * FDIM + c]);
    atomicMax(gmax + c, __float_as_uint(m));  // non-negative floats: uint order == float order
}

// ---------------- out = gmax @ Wlin + blin ----------------
__global__ void final_linear_kernel(const float* __restrict__ g, const float* __restrict__ Wlin,
                                    const float* __restrict__ blin, float* __restrict__ out) {
    int col = threadIdx.x;               // 128 threads
    float s = blin[col];
    #pragma unroll 8
    for (int k = 0; k < FDIM; ++k) s = fmaf(g[k], Wlin[(size_t)k * FDIM + col], s);
    out[col] = s;
}

extern "C" void kernel_launch(void* const* d_in, const int* in_sizes, int n_in,
                              void* d_out, int out_size, void* d_ws, size_t ws_size,
                              hipStream_t stream) {
    const float* x    = (const float*)d_in[0];
    const int*   ei   = (const int*)  d_in[1];
    const float* W1   = (const float*)d_in[2];
    const float* b1   = (const float*)d_in[3];
    const float* W2   = (const float*)d_in[4];
    const float* b2   = (const float*)d_in[5];
    const float* W3   = (const float*)d_in[6];
    const float* b3   = (const float*)d_in[7];
    const float* Wlin = (const float*)d_in[8];
    const float* blin = (const float*)d_in[9];

    const int N = in_sizes[0] / FDIM;
    const int E = in_sizes[1] / 2;
    const int* srcp = ei;
    const int* dstp = ei + E;

    // workspace layout: dis[N] | gmax[128] | P[N*128] | Q[N*128]
    float* ws  = (float*)d_ws;
    float* dis = ws;
    unsigned* gmax = (unsigned*)(ws + N);
    size_t o = ((size_t)N + FDIM + 3) & ~(size_t)3;
    float* P = ws + o;
    float* Q = P + (size_t)N * FDIM;

    // --- normalization coefficients ---
    deg_init_kernel<<<(N + 255) / 256, 256, 0, stream>>>(dis, N);
    deg_count_kernel<<<(E + 255) / 256, 256, 0, stream>>>(dstp, dis, E);
    deg_rsqrt_kernel<<<(N + 255) / 256, 256, 0, stream>>>(dis, N);

    const int mtiles = (N + 15) / 16;
    const int gemm_blocks = (mtiles + 7) / 8;
    const int nt32  = N * 32;
    const int nc    = N * FDIM;
    const long long e32 = (long long)E * 32;
    const unsigned edge_blocks = (unsigned)((e32 + 255) / 256);

    auto run_layer = [&](const float* hin, const float* W, const float* b,
                         float* T, float* agg, float* hout) {
        gemm_bf16_kernel<<<gemm_blocks, 256, 0, stream>>>(hin, W, T, N);
        self_init_kernel<<<(nt32 + 255) / 256, 256, 0, stream>>>(T, dis, agg, N);
        edge_scatter_kernel<<<edge_blocks, 256, 0, stream>>>(srcp, dstp, dis, T, agg, E);
        bias_relu_kernel<<<(nc + 255) / 256, 256, 0, stream>>>(agg, b, hout, nc);
    };

    // ping-pong buffers (t is dead by the time hout overwrites it)
    run_layer(x, W1, b1, /*T=*/P, /*agg=*/Q, /*hout=*/P);
    run_layer(P, W2, b2, /*T=*/Q, /*agg=*/P, /*hout=*/Q);
    run_layer(Q, W3, b3, /*T=*/P, /*agg=*/Q, /*hout=*/P);

    pool_init_kernel<<<1, 128, 0, stream>>>(gmax);
    pool_max_kernel<<<(N + 127) / 128, 128, 0, stream>>>(P, gmax, N);
    final_linear_kernel<<<1, 128, 0, stream>>>((const float*)gmax, Wlin, blin, (float*)d_out);
}
